// GAT_15633680957583
// MI455X (gfx1250) — matrix operand order repaired and checked by reading, compiled, not run on hardware
//
#include <hip/hip_runtime.h>
#include <hip/hip_bf16.h>
#include <math.h>

typedef __attribute__((ext_vector_type(16))) _Float16 v16h;
typedef __attribute__((ext_vector_type(8)))  _Float16 v8h;
typedef __attribute__((ext_vector_type(8)))  float    v8f;

#define N_NODES 50000
#define N_EDGES 800000
#define HIDC    16
#define HEADS6  6
#define HW96    96
#define NEG_SLOPE 0.2f

static_assert(N_NODES % 16 == 0, "M-tiling assumes N_NODES multiple of 16");

// ---------------------------------------------------------------------------
// WMMA GEMM: C[N_NODES x KOUT] = A[N_NODES x KIN] @ W[KIN x KOUT] (+biases)
// f32 inputs converted to f16 in LDS, f32 accumulate via v_wmma_f32_16x16x32_f16.
// One wave per 16x16 output tile; block = 1 M-tile x (KOUT/16) waves.
// W staged into LDS pre-swizzled into B-fragment order so each lane's 16 halfs
// are contiguous -> 2x ds_load_b128 per WMMA (A likewise).
// BMODE: 0 = no bias, 2 = bias1+bias2 added to every output column.
// ---------------------------------------------------------------------------
template<int KIN, int KOUT, int BMODE>
__global__ __launch_bounds__(32*(KOUT/16))
void gemm_wmma(const float* __restrict__ A, const float* __restrict__ W,
               const float* __restrict__ bias1, const float* __restrict__ bias2,
               float* __restrict__ C)
{
    __shared__ __align__(16) _Float16 Alds[16 * KIN];
    __shared__ __align__(16) _Float16 Wlds[KIN * KOUT];   // fragment-swizzled

    const int tid = threadIdx.x;
    const int nth = blockDim.x;
    const int m0  = blockIdx.x * 16;

    // Stage A tile (16 x KIN) into LDS as f16, row-major (grid tiles M exactly)
    for (int i = tid; i < 16 * KIN; i += nth) {
        int r = i / KIN, c = i % KIN;
        Alds[i] = (_Float16)A[(long long)(m0 + r) * KIN + c];
    }
    // Stage W (KIN x KOUT) into LDS swizzled to B-fragment order:
    //   element (k, c) -> [ (k/32)*2 + (k%32)/16 ][ c ][ k%16 ]
    for (int i = tid; i < KIN * KOUT; i += nth) {
        int k = i / KOUT, c = i % KOUT;
        int kt  = k >> 5;
        int rem = k & 31;
        int bhi = rem >> 4;
        int j   = rem & 15;
        Wlds[((kt * 2 + bhi) * KOUT + c) * 16 + j] = (_Float16)W[i];
    }
    __syncthreads();

    const int wave = tid >> 5;
    const int lane = tid & 31;
    const int n0   = wave * 16;      // column tile handled by this wave
    const int col  = lane & 15;
    const int hi   = lane >> 4;      // half-wave select
    const int am   = lane & 15;      // A-fragment row

    v8f acc = {};
    #pragma unroll
    for (int k0 = 0; k0 < KIN; k0 += 32) {
        // A fragment: lane holds K {kbase..kbase+15}, kbase = k0+16*hi
        const v8h* ap = (const v8h*)&Alds[am * KIN + k0 + hi * 16];
        v8h a0 = ap[0];
        v8h a1 = ap[1];                      // +8 halfs
        v16h a = __builtin_shufflevector(a0, a1, 0,1,2,3,4,5,6,7,8,9,10,11,12,13,14,15);
        // B fragment: 16 contiguous halfs in swizzled Wlds
        const v8h* bp = (const v8h*)&Wlds[(((k0 >> 5) * 2 + hi) * KOUT + (n0 + col)) * 16];
        v8h b0 = bp[0];
        v8h b1 = bp[1];
        v16h b = __builtin_shufflevector(b0, b1, 0,1,2,3,4,5,6,7,8,9,10,11,12,13,14,15);

        acc = __builtin_amdgcn_wmma_f32_16x16x32_f16(
            /*neg_a=*/false, a, /*neg_b=*/false, b,
            /*c_mod=*/(short)0, acc, /*reuse_a=*/false, /*reuse_b=*/false);
    }

    float badd = 0.0f;
    if (BMODE == 2) badd = bias1[n0 + col] + bias2[n0 + col];

    // C layout: VGPR r -> row r + 8*hi, col = lane&15 ; no row guard (exact tiling)
    float* cp = C + (long long)(m0 + hi * 8) * KOUT + n0 + col;
    #pragma unroll
    for (int r = 0; r < 8; ++r)
        cp[(long long)r * KOUT] = acc[r] + badd;
}

// ---------------------------------------------------------------------------
// Edge phase kernels (L2-resident gather/scatter, f32 atomics)
// ---------------------------------------------------------------------------
__device__ __forceinline__ void atomicMaxF(float* addr, float v) {
    if (v >= 0.0f) atomicMax((int*)addr, __float_as_int(v));
    else           atomicMin((unsigned int*)addr, __float_as_uint(v));
}

__global__ void init_md(float* m, float* denom, int n) {
    int t = blockIdx.x * blockDim.x + threadIdx.x;
    if (t < n) { m[t] = -__builtin_inff(); denom[t] = 0.0f; }
}

template<int H>
__global__ void edge_scores(const float* __restrict__ xl, const float* __restrict__ xr,
                            const int* __restrict__ src, const int* __restrict__ dst,
                            const float* __restrict__ att,
                            float* __restrict__ e, float* __restrict__ m)
{
    long long t = (long long)blockIdx.x * blockDim.x + threadIdx.x;
    if (t >= (long long)N_EDGES * H) return;
    int edge = (int)(t / H), h = (int)(t % H);
    int s = src[edge], d = dst[edge];
    const float4* pl = (const float4*)(xl + (long long)s * (H * HIDC) + h * HIDC);
    const float4* pr = (const float4*)(xr + (long long)d * (H * HIDC) + h * HIDC);
    const float4* pa = (const float4*)(att + h * HIDC);
    float acc = 0.0f;
    #pragma unroll
    for (int q = 0; q < 4; ++q) {
        float4 vl = pl[q], vr = pr[q], va = pa[q];
        float v0 = vl.x + vr.x, v1 = vl.y + vr.y, v2 = vl.z + vr.z, v3 = vl.w + vr.w;
        v0 = (v0 > 0.0f) ? v0 : NEG_SLOPE * v0;
        v1 = (v1 > 0.0f) ? v1 : NEG_SLOPE * v1;
        v2 = (v2 > 0.0f) ? v2 : NEG_SLOPE * v2;
        v3 = (v3 > 0.0f) ? v3 : NEG_SLOPE * v3;
        acc += v0 * va.x + v1 * va.y + v2 * va.z + v3 * va.w;
    }
    e[t] = acc;
    atomicMaxF(&m[(long long)d * H + h], acc);
}

template<int H>
__global__ void edge_exp(const int* __restrict__ dst, const float* __restrict__ m,
                         float* __restrict__ e, float* __restrict__ denom)
{
    long long t = (long long)blockIdx.x * blockDim.x + threadIdx.x;
    if (t >= (long long)N_EDGES * H) return;
    int edge = (int)(t / H), h = (int)(t % H);
    int d = dst[edge];
    float ee = __expf(e[t] - m[(long long)d * H + h]);
    e[t] = ee;
    atomicAdd(&denom[(long long)d * H + h], ee);
}

template<int H>
__global__ void edge_aggr(const float* __restrict__ xl,
                          const int* __restrict__ src, const int* __restrict__ dst,
                          const float* __restrict__ e, const float* __restrict__ denom,
                          float* __restrict__ agg)
{
    long long t = (long long)blockIdx.x * blockDim.x + threadIdx.x;
    if (t >= (long long)N_EDGES * H) return;
    int edge = (int)(t / H), h = (int)(t % H);
    int s = src[edge], d = dst[edge];
    float alpha = e[t] / (denom[(long long)d * H + h] + 1e-16f);
    const float4* pl = (const float4*)(xl + (long long)s * (H * HIDC) + h * HIDC);
    float*        po = agg + (long long)d * (H * HIDC) + h * HIDC;
    #pragma unroll
    for (int q = 0; q < 4; ++q) {
        float4 vl = pl[q];
        atomicAdd(&po[4 * q + 0], alpha * vl.x);
        atomicAdd(&po[4 * q + 1], alpha * vl.y);
        atomicAdd(&po[4 * q + 2], alpha * vl.z);
        atomicAdd(&po[4 * q + 3], alpha * vl.w);
    }
}

// Final MLP: lin1(16->16) + ELU + lin2(16->1)
__global__ void final_mlp(const float* __restrict__ x4,
                          const float* __restrict__ W1, const float* __restrict__ b1,
                          const float* __restrict__ W2, const float* __restrict__ b2,
                          float* __restrict__ out, int n)
{
    int t = blockIdx.x * blockDim.x + threadIdx.x;
    if (t >= n) return;
    float xin[16];
    const float4* px = (const float4*)(x4 + (long long)t * 16);
    #pragma unroll
    for (int q = 0; q < 4; ++q) {
        float4 v = px[q];
        xin[4 * q + 0] = v.x; xin[4 * q + 1] = v.y;
        xin[4 * q + 2] = v.z; xin[4 * q + 3] = v.w;
    }
    float o = b2[0];
    #pragma unroll
    for (int j = 0; j < 16; ++j) {
        float s = b1[j];
        #pragma unroll
        for (int c = 0; c < 16; ++c) s += xin[c] * W1[c * 16 + j];
        s = (s > 0.0f) ? s : expm1f(s);   // ELU
        o += s * W2[j];
    }
    out[t] = o;
}

// ---------------------------------------------------------------------------
// Host-side layer driver
// ---------------------------------------------------------------------------
template<int KIN, int KOUT, int H>
static void run_layer(const float* xin, const float* Wl, const float* Wr,
                      const float* att, const float* cb,
                      const float* Sw, const float* Sb,
                      const int* src, const int* dst,
                      float* xl, float* xr, float* ebuf, float* mbuf, float* dbuf,
                      float* aggout, hipStream_t stream)
{
    dim3 blk(32 * (KOUT / 16));
    dim3 grd(N_NODES / 16);
    gemm_wmma<KIN, KOUT, 0><<<grd, blk, 0, stream>>>(xin, Wl, nullptr, nullptr, xl);
    gemm_wmma<KIN, KOUT, 0><<<grd, blk, 0, stream>>>(xin, Wr, nullptr, nullptr, xr);
    // agg initialized with skip(x)+skip_bias+conv_bias; edge atomics accumulate on top
    gemm_wmma<KIN, KOUT, 2><<<grd, blk, 0, stream>>>(xin, Sw, Sb, cb, aggout);

    int nmd = N_NODES * H;
    init_md<<<(nmd + 255) / 256, 256, 0, stream>>>(mbuf, dbuf, nmd);

    long long tot = (long long)N_EDGES * H;
    int eblocks = (int)((tot + 255) / 256);
    edge_scores<H><<<eblocks, 256, 0, stream>>>(xl, xr, src, dst, att, ebuf, mbuf);
    edge_exp<H>   <<<eblocks, 256, 0, stream>>>(dst, mbuf, ebuf, dbuf);
    edge_aggr<H>  <<<eblocks, 256, 0, stream>>>(xl, src, dst, ebuf, dbuf, aggout);
}

extern "C" void kernel_launch(void* const* d_in, const int* in_sizes, int n_in,
                              void* d_out, int out_size, void* d_ws, size_t ws_size,
                              hipStream_t stream)
{
    (void)in_sizes; (void)n_in; (void)out_size; (void)ws_size;

    const float* x    = (const float*)d_in[0];
    const int*   ei   = (const int*)d_in[1];
    const int*   src  = ei;
    const int*   dst  = ei + N_EDGES;

    const float* c1Wl = (const float*)d_in[2];
    const float* c1Wr = (const float*)d_in[3];
    const float* c1at = (const float*)d_in[4];
    const float* c1b  = (const float*)d_in[5];
    const float* c2Wl = (const float*)d_in[6];
    const float* c2Wr = (const float*)d_in[7];
    const float* c2at = (const float*)d_in[8];
    const float* c2b  = (const float*)d_in[9];
    const float* c3Wl = (const float*)d_in[10];
    const float* c3Wr = (const float*)d_in[11];
    const float* c3at = (const float*)d_in[12];
    const float* c3b  = (const float*)d_in[13];
    const float* c4Wl = (const float*)d_in[14];
    const float* c4Wr = (const float*)d_in[15];
    const float* c4at = (const float*)d_in[16];
    const float* c4b  = (const float*)d_in[17];
    const float* s1w  = (const float*)d_in[18];
    const float* s1b  = (const float*)d_in[19];
    const float* s2w  = (const float*)d_in[20];
    const float* s2b  = (const float*)d_in[21];
    const float* s3w  = (const float*)d_in[22];
    const float* s3b  = (const float*)d_in[23];
    const float* s4w  = (const float*)d_in[24];
    const float* s4b  = (const float*)d_in[25];
    const float* l1w  = (const float*)d_in[26];
    const float* l1b  = (const float*)d_in[27];
    const float* l2w  = (const float*)d_in[28];
    const float* l2b  = (const float*)d_in[29];

    float* ws = (float*)d_ws;
    size_t off = 0;
    float* bufA = ws + off; off += (size_t)N_NODES * HW96;
    float* bufB = ws + off; off += (size_t)N_NODES * HW96;
    float* xl   = ws + off; off += (size_t)N_NODES * HW96;
    float* xr   = ws + off; off += (size_t)N_NODES * HW96;
    float* ebuf = ws + off; off += (size_t)N_EDGES * HEADS6;
    float* mbuf = ws + off; off += (size_t)N_NODES * HEADS6;
    float* dbuf = ws + off; off += (size_t)N_NODES * HEADS6;
    float* buf4 = ws + off; off += (size_t)N_NODES * HIDC;

    // Layer 1: 128 -> 6x16, out in bufA
    run_layer<128, 96, 6>(x,    c1Wl, c1Wr, c1at, c1b, s1w, s1b,
                          src, dst, xl, xr, ebuf, mbuf, dbuf, bufA, stream);
    // Layer 2: 96 -> 6x16, out in bufB
    run_layer<96, 96, 6>(bufA, c2Wl, c2Wr, c2at, c2b, s2w, s2b,
                         src, dst, xl, xr, ebuf, mbuf, dbuf, bufB, stream);
    // Layer 3: 96 -> 6x16, out in bufA
    run_layer<96, 96, 6>(bufB, c3Wl, c3Wr, c3at, c3b, s3w, s3b,
                         src, dst, xl, xr, ebuf, mbuf, dbuf, bufA, stream);
    // Layer 4: 96 -> 1x16, out in buf4
    run_layer<96, 16, 1>(bufA, c4Wl, c4Wr, c4at, c4b, s4w, s4b,
                         src, dst, xl, xr, ebuf, mbuf, dbuf, buf4, stream);

    // Final MLP
    final_mlp<<<(N_NODES + 255) / 256, 256, 0, stream>>>(
        buf4, l1w, l1b, l2w, l2b, (float*)d_out, N_NODES);
}
